// CascadedNetworkObj_8632884264969
// MI455X (gfx1250) — matrix operand order repaired and checked by reading, not compiled, not run on hardware
//
#include <hip/hip_runtime.h>
#include <hip/hip_bf16.h>
#include <math.h>

// ---------------------------------------------------------------------------
// MI455X (gfx1250) implementation notes:
//  * wave32 only; WMMA 16x16x32 bf16 (f32 accumulate) for all matmuls.
//  * SA weights pre-padded to bf16 [Nout][Kpad32] so every WMMA B-fragment is
//    two unconditional global_load_b128 (no exec-masked scalar loads).
//  * Grouped SA-MLPs fused per (b,s) group: gather -> 3x (GEMM+BN+ReLU) in
//    LDS ping-pong bf16 buffers -> max-pool. Activations never touch HBM.
//  * Stage FC heads: M=16 (batch), wave-split-K; the streaming fp32 weights
//    (64 MB / 16 MB, the only real HBM term: ~4 us at 23.3 TB/s) are staged
//    coalesced into a transposed LDS tile, B-fragments = 2x ds_load_b128.
//    Ragged K handled by a wave-uniform branch (EXEC all-1s around WMMA).
//  * FPS / ball query / top-k are latency-bound scalar/VALU kernels.
// ---------------------------------------------------------------------------

typedef __bf16 bf16_t;
typedef __attribute__((ext_vector_type(16))) __bf16 v16bf;
typedef __attribute__((ext_vector_type(8)))  float  v8f;

#define BN_EPS 1e-5f

// ================= weight pre-pad: (Nout,Kc) f32 -> (Nout,Kpad) bf16 =======
__global__ void pad_weights_kernel(const float* __restrict__ w, int Kc,
                                   int Kpad, int total,
                                   bf16_t* __restrict__ out) {
  const int i = blockIdx.x * blockDim.x + threadIdx.x;
  if (i >= total) return;
  const int o = i / Kpad, k = i - o * Kpad;
  out[i] = (bf16_t)((k < Kc) ? w[(size_t)o * Kc + k] : 0.f);
}

// ============================ FPS ==========================================
// one block per batch; thread t owns point t; npoint dependent argmax rounds.
__global__ void fps_kernel(const float* __restrict__ xyz, int stride, int N,
                           int npoint, int* __restrict__ fidx) {
  __shared__ float sd[1024];
  __shared__ int   si[1024];
  __shared__ float cc[3];
  const int b = blockIdx.x, t = threadIdx.x;
  const bool valid = t < N;
  float px = 0.f, py = 0.f, pz = 0.f;
  if (valid) {
    const float* p = xyz + ((size_t)b * N + t) * stride;
    px = p[0]; py = p[1]; pz = p[2];
  }
  float dist = 1e10f;
  int far = 0;
  for (int it = 0; it < npoint; ++it) {
    if (t == 0) {
      fidx[b * npoint + it] = far;
      const float* p = xyz + ((size_t)b * N + far) * stride;
      cc[0] = p[0]; cc[1] = p[1]; cc[2] = p[2];
    }
    __syncthreads();
    const float dx = px - cc[0], dy = py - cc[1], dz = pz - cc[2];
    if (valid) dist = fminf(dist, dx * dx + dy * dy + dz * dz);
    sd[t] = valid ? dist : -1.f;
    si[t] = t;
    __syncthreads();
    for (int off = 512; off > 0; off >>= 1) {
      if (t < off) {
        const float v1 = sd[t], v2 = sd[t + off];
        const int   i1 = si[t], i2 = si[t + off];
        if (v2 > v1 || (v2 == v1 && i2 < i1)) { sd[t] = v2; si[t] = i2; }
      }
      __syncthreads();
    }
    far = si[0];
    __syncthreads();
  }
}

// ====================== gather sampled centroids ===========================
__global__ void gather_newxyz_kernel(const float* __restrict__ src, int stride,
                                     int N, const int* __restrict__ fidx,
                                     int total, float* __restrict__ dst, int S) {
  const int i = blockIdx.x * blockDim.x + threadIdx.x;
  if (i >= total) return;
  const int b = i / S;
  const int idx = fidx[i];
  const float* p = src + ((size_t)b * N + idx) * stride;
  dst[(size_t)i * 3 + 0] = p[0];
  dst[(size_t)i * 3 + 1] = p[1];
  dst[(size_t)i * 3 + 2] = p[2];
}

// ============================ ball query ===================================
// first K in-radius indices in ascending index order, pad with first hit.
__global__ void ball_query_kernel(const float* __restrict__ xyz, int stride,
                                  int N, const float* __restrict__ newxyz,
                                  int total, int S, float r2,
                                  int* __restrict__ gidx) {
  const int i = blockIdx.x * blockDim.x + threadIdx.x;
  if (i >= total) return;
  const int b = i / S;
  const float* nx = newxyz + (size_t)i * 3;
  const float cx = nx[0], cy = nx[1], cz = nx[2];
  int* g = gidx + (size_t)i * 64;
  int cnt = 0, first = 0;
  bool hasf = false;
  for (int p = 0; p < N && cnt < 64; ++p) {
    const float* q = xyz + ((size_t)b * N + p) * stride;
    const float dx = q[0] - cx, dy = q[1] - cy, dz = q[2] - cz;
    if (dx * dx + dy * dy + dz * dz <= r2) {
      if (!hasf) { first = p; hasf = true; }
      g[cnt++] = p;
    }
  }
  for (; cnt < 64; ++cnt) g[cnt] = first;
}

// ===================== fused SA MLP (WMMA bf16) ============================
// One 128-thread block (4 waves) per (b,s) group; 64 grouped points.
// A fragment (16-bit A 16x32, contiguous k order used by this code):
//   lanes 0-15:  row=lane,    k=k0..k0+15
//   lanes 16-31: row=lane-16, k=k0+16..k0+31
// B fragment (32x16): lanes 0-15: n=lane, k=k0..k0+15;
//   lanes 16-31: n=lane-16, k=k0+16..k0+31.
__device__ __forceinline__ void mlp_layer(
    const bf16_t* __restrict__ Xs, int CpadS, bf16_t* __restrict__ Xd, int CpadD,
    const bf16_t* __restrict__ wp /* [Nout][Kpad] bf16, zero-padded */,
    const float* __restrict__ bias,
    const float* __restrict__ gam, const float* __restrict__ bet,
    const float* __restrict__ mean, const float* __restrict__ var,
    int Kpad, int Nout, int lane, int wave) {
  const int r0    = wave * 16;
  const int arow  = r0 + (lane & 15);
  const int akoff = (lane < 16) ? 0 : 16;
  const int bko   = (lane < 16) ? 0 : 16;
  const bf16_t* xr = Xs + (size_t)arow * CpadS;
  for (int n0 = 0; n0 < Nout; n0 += 16) {
    v8f acc = {0.f, 0.f, 0.f, 0.f, 0.f, 0.f, 0.f, 0.f};
    const int o = n0 + (lane & 15);
    const bf16_t* wr = wp + (size_t)o * Kpad;
    for (int k0 = 0; k0 < Kpad; k0 += 32) {
      v16bf af, bfg;
      const int ka = k0 + akoff;
#pragma unroll
      for (int j = 0; j < 8; ++j) {
        af[j]     = xr[ka + j];
        af[8 + j] = xr[ka + 8 + j];
      }
      const bf16_t* wb = wr + k0 + bko;
#pragma unroll
      for (int j = 0; j < 16; ++j) bfg[j] = wb[j];
      acc = __builtin_amdgcn_wmma_f32_16x16x32_bf16(
          false, af, false, bfg, (short)0, acc, false, false);
    }
    const float scale = gam[o] * rsqrtf(var[o] + BN_EPS);
    const float shift = bet[o] + (bias[o] - mean[o]) * scale;
#pragma unroll
    for (int vv = 0; vv < 8; ++vv) {
      const int m = vv + ((lane < 16) ? 0 : 8);
      float val = acc[vv] * scale + shift;
      val = fmaxf(val, 0.f);
      Xd[(size_t)(r0 + m) * CpadD + o] = (bf16_t)val;
    }
  }
}

__global__ void sa_mlp_kernel(
    const float* __restrict__ xyz, int xyz_stride, int N,
    const float* __restrict__ pts, int pts_cf, int pts_stride, int ptsC,
    const float* __restrict__ newxyz, const int* __restrict__ gidx,
    int S, int C_in, int Cpad0, int Cpad1,
    const bf16_t* w0p, int Kp0, const float* b0, const float* g0,
    const float* be0, const float* m0, const float* v0, int C1,
    const bf16_t* w1p, const float* b1, const float* g1,
    const float* be1, const float* m1, const float* v1, int C2,
    const bf16_t* w2p, const float* b2, const float* g2,
    const float* be2, const float* m2, const float* v2, int C3,
    float* __restrict__ lpts) {
  extern __shared__ char smem_raw[];
  bf16_t* X0 = (bf16_t*)smem_raw;
  bf16_t* X1 = X0 + 64 * Cpad0;
  const int tid = threadIdx.x, lane = tid & 31, wave = tid >> 5;
  const int bs = blockIdx.x;
  const int b = bs / S, s = bs - b * S;
  const int* gi = gidx + (size_t)bs * 64;
  const float* nx = newxyz + (size_t)bs * 3;
  const float cx = nx[0], cy = nx[1], cz = nx[2];

  // gather grouped features: [rel_xyz | pts], zero-pad to Cpad0
  for (int e = tid; e < 64 * Cpad0; e += 128) {
    const int row = e / Cpad0, c = e - row * Cpad0;
    float val = 0.f;
    if (c < C_in) {
      const int idx = gi[row];
      if (c < 3) {
        val = xyz[((size_t)b * N + idx) * xyz_stride + c] -
              (c == 0 ? cx : (c == 1 ? cy : cz));
      } else {
        const int pc = c - 3;
        val = pts_cf ? pts[((size_t)b * ptsC + pc) * N + idx]
                     : pts[((size_t)b * N + idx) * pts_stride + pc];
      }
    }
    X0[e] = (bf16_t)val;
  }
  __syncthreads();
  mlp_layer(X0, Cpad0, X1, Cpad1, w0p, b0, g0, be0, m0, v0, Kp0, C1, lane, wave);
  __syncthreads();
  mlp_layer(X1, Cpad1, X0, Cpad0, w1p, b1, g1, be1, m1, v1, C1, C2, lane, wave);
  __syncthreads();
  mlp_layer(X0, Cpad0, X1, Cpad1, w2p, b2, g2, be2, m2, v2, C2, C3, lane, wave);
  __syncthreads();
  // max-pool over the 64 grouped points -> lpts[b][c][s] (channel-first)
  for (int c = tid; c < C3; c += 128) {
    float mx = (float)X1[c];
    for (int r = 1; r < 64; ++r)
      mx = fmaxf(mx, (float)X1[(size_t)r * Cpad1 + c]);
    lpts[((size_t)b * C3 + c) * S + s] = mx;
  }
}

// ====================== stage FC heads (WMMA bf16) =========================
// M = 16 (batch), one 16-wide N tile per block, 8 waves split a 256-k chunk.
// W (K x N, f32) is staged coalesced into a transposed LDS tile [n][k] so
// B-fragments are two unconditional ds_load_b128. K must be a multiple of 32.
// mode 0: out = relu(bn(acc + bias));  mode 1: out = acc + bias
#define WLDS_STRIDE 264  // 256 + 8 bf16 pad: spreads LDS banks
__global__ void stage_gemm_kernel(const float* __restrict__ A, int K,
                                  const float* __restrict__ W, int N,
                                  const float* __restrict__ bias,
                                  const float* __restrict__ gam,
                                  const float* __restrict__ bet,
                                  const float* __restrict__ mean,
                                  const float* __restrict__ var,
                                  int mode, float* __restrict__ out) {
  __shared__ bf16_t wlds[16 * WLDS_STRIDE];
  __shared__ float red[8 * 256];
  const int tid = threadIdx.x, lane = tid & 31, wave = tid >> 5;
  const int n0 = blockIdx.x * 16;
  const int arow = lane & 15;
  const int akoff = (lane < 16) ? 0 : 16;
  const int bko = (lane < 16) ? 0 : 16;
  const int kl = wave * 32;                  // wave's k slice in each chunk
  const int snn = tid & 15, skr = tid >> 4;  // staging coords
  v8f acc = {0.f, 0.f, 0.f, 0.f, 0.f, 0.f, 0.f, 0.f};
  const float* ar = A + (size_t)arow * K;
  const bf16_t* brow = wlds + (size_t)(lane & 15) * WLDS_STRIDE + bko;

  for (int k0c = 0; k0c < K; k0c += 256) {
    // ---- stage 256x16 W tile, coalesced over n, transposed into LDS ----
#pragma unroll
    for (int p = 0; p < 16; ++p) {
      const int kk = p * 16 + skr;
      const int k = k0c + kk;
      const float v =
          (k < K && (n0 + snn) < N) ? W[(size_t)k * N + n0 + snn] : 0.f;
      wlds[snn * WLDS_STRIDE + kk] = (bf16_t)v;
    }
    __syncthreads();
    if (k0c + kl < K) {  // wave-uniform: EXEC stays all-1s for WMMA
      v16bf af, bfg;
      const float* a1 = ar + k0c + kl + akoff;
#pragma unroll
      for (int j = 0; j < 8; ++j) {
        af[j]     = (bf16_t)a1[j];
        af[8 + j] = (bf16_t)a1[8 + j];
      }
      const bf16_t* wb = brow + kl;
#pragma unroll
      for (int j = 0; j < 16; ++j) bfg[j] = wb[j];
      acc = __builtin_amdgcn_wmma_f32_16x16x32_bf16(
          false, af, false, bfg, (short)0, acc, false, false);
    }
    __syncthreads();
  }
#pragma unroll
  for (int vv = 0; vv < 8; ++vv) {
    const int m = vv + ((lane < 16) ? 0 : 8);
    red[wave * 256 + m * 16 + (lane & 15)] = acc[vv];
  }
  __syncthreads();
  const int m = tid >> 4, nn = tid & 15;
  const int nc = n0 + nn;
  if (nc < N) {
    float sum = 0.f;
#pragma unroll
    for (int w = 0; w < 8; ++w) sum += red[w * 256 + m * 16 + nn];
    sum += bias[nc];
    if (mode == 0) {
      const float sc = gam[nc] * rsqrtf(var[nc] + BN_EPS);
      sum = (sum - mean[nc]) * sc + bet[nc];
      sum = fmaxf(sum, 0.f);
    }
    out[(size_t)m * N + nc] = sum;
  }
}

// =================== joints = pca_mean + e2 @ pca_coeff ====================
__global__ void joints_kernel(const float* __restrict__ e2,
                              const float* __restrict__ pmean,
                              const float* __restrict__ coeff,
                              float* __restrict__ joints) {
  const int i = blockIdx.x * blockDim.x + threadIdx.x;
  if (i >= 16 * 42) return;
  const int b = i / 42, j = i - b * 42;
  float s = pmean[j];
  for (int k = 0; k < 42; ++k) s += e2[b * 42 + k] * coeff[k * 42 + j];
  joints[i] = s;
}

// =========== per-joint kNN (k=64) gather of 6-dim point features ===========
__global__ void topk_group_kernel(const float* __restrict__ xyz6,
                                  const float* __restrict__ joints,
                                  float* __restrict__ grouped) {
  __shared__ float d[1024];
  __shared__ int   sel[64];
  __shared__ float rv[256];
  __shared__ int   ri[256];
  const int bj = blockIdx.x;
  const int b = bj / 14, j = bj - b * 14;
  const int t = threadIdx.x;
  const float jx = joints[b * 42 + j * 3 + 0];
  const float jy = joints[b * 42 + j * 3 + 1];
  const float jz = joints[b * 42 + j * 3 + 2];
  for (int p = t; p < 1024; p += 256) {
    const float* q = xyz6 + ((size_t)b * 1024 + p) * 6;
    const float dx = q[0] - jx, dy = q[1] - jy, dz = q[2] - jz;
    d[p] = dx * dx + dy * dy + dz * dz;
  }
  __syncthreads();
  for (int it = 0; it < 64; ++it) {
    float bv = 1e30f;
    int bi = 0x7fffffff;
    for (int p = t; p < 1024; p += 256) {
      const float v = d[p];
      if (v < bv) { bv = v; bi = p; }
    }
    rv[t] = bv; ri[t] = bi;
    __syncthreads();
    for (int off = 128; off > 0; off >>= 1) {
      if (t < off) {
        if (rv[t + off] < rv[t] ||
            (rv[t + off] == rv[t] && ri[t + off] < ri[t])) {
          rv[t] = rv[t + off]; ri[t] = ri[t + off];
        }
      }
      __syncthreads();
    }
    if (t == 0) { sel[it] = ri[0]; d[ri[0]] = 1e30f; }
    __syncthreads();
  }
  for (int e = t; e < 64 * 6; e += 256) {
    const int g = e / 6, c = e - g * 6;
    grouped[((size_t)b * 896 + j * 64 + g) * 6 + c] =
        xyz6[((size_t)b * 1024 + sel[g]) * 6 + c];
  }
}

// ===========================================================================
extern "C" void kernel_launch(void* const* d_in, const int* in_sizes, int n_in,
                              void* d_out, int out_size, void* d_ws,
                              size_t ws_size, hipStream_t stream) {
  (void)in_sizes; (void)n_in; (void)out_size; (void)ws_size;
  auto F = [&](int i) { return (const float*)d_in[i]; };

  // jax pytree (sorted-key) flattening of setup_inputs():
  const float* pca_coeff = F(0);
  const float* pca_mean  = F(1);
  const float *s1b0 = F(2), *s1b1 = F(3), *s1b2 = F(4);
  const float *s1be0 = F(5),  *s1g0 = F(6),  *s1m0 = F(7),  *s1v0 = F(8);
  const float *s1be1 = F(9),  *s1g1 = F(10), *s1m1 = F(11), *s1v1 = F(12);
  const float *s1be2 = F(13), *s1g2 = F(14), *s1m2 = F(15), *s1v2 = F(16);
  const float *s1w0 = F(17), *s1w1 = F(18), *s1w2 = F(19);
  const float *s2b0 = F(20), *s2b1 = F(21), *s2b2 = F(22);
  const float *s2be0 = F(23), *s2g0 = F(24), *s2m0 = F(25), *s2v0 = F(26);
  const float *s2be1 = F(27), *s2g1 = F(28), *s2m1 = F(29), *s2v1 = F(30);
  const float *s2be2 = F(31), *s2g2 = F(32), *s2m2 = F(33), *s2v2 = F(34);
  const float *s2w0 = F(35), *s2w1 = F(36), *s2w2 = F(37);
  const float *s3b0 = F(38), *s3b1 = F(39), *s3b2 = F(40);
  const float *s3be0 = F(41), *s3g0 = F(42), *s3m0 = F(43), *s3v0 = F(44);
  const float *s3be1 = F(45), *s3g1 = F(46), *s3m1 = F(47), *s3v1 = F(48);
  const float *s3be2 = F(49), *s3g2 = F(50), *s3m2 = F(51), *s3v2 = F(52);
  const float *s3w0 = F(53), *s3w1 = F(54), *s3w2 = F(55);
  const float *st1be = F(56), *st1g = F(57), *st1m = F(58), *st1v = F(59);
  const float *st1l1b = F(60), *st1l1w = F(61), *st1l2b = F(62), *st1l2w = F(63);
  const float *st2be = F(64), *st2g = F(65), *st2m = F(66), *st2v = F(67);
  const float *st2l1b = F(68), *st2l1w = F(69), *st2l2b = F(70), *st2l2w = F(71);
  const float* xyz6 = F(72);  // (16, 1024, 6)

  float* out = (float*)d_out;
  float* e1 = out;          // (16,42)
  float* e2 = out + 672;    // (16,42)
  float* e3 = out + 1344;   // (16,42)

  // ---- workspace bump allocator ----
  char* ws = (char*)d_ws;
  size_t off = 0;
  auto alloc = [&](size_t bytes) {
    void* p = ws + off;
    off = (off + bytes + 255) & ~(size_t)255;
    return p;
  };
  const int B = 16;
  int*   fidx1   = (int*)alloc((size_t)B * 512 * 4);
  float* newxyz1 = (float*)alloc((size_t)B * 512 * 3 * 4);
  int*   gidx1   = (int*)alloc((size_t)B * 512 * 64 * 4);
  float* l1_pts  = (float*)alloc((size_t)B * 128 * 512 * 4);
  float* h1      = (float*)alloc((size_t)B * 256 * 4);
  int*   fidx2   = (int*)alloc((size_t)B * 128 * 4);
  float* newxyz2 = (float*)alloc((size_t)B * 128 * 3 * 4);
  int*   gidx2   = (int*)alloc((size_t)B * 128 * 64 * 4);
  float* l2_pts  = (float*)alloc((size_t)B * 256 * 128 * 4);
  float* h2      = (float*)alloc((size_t)B * 128 * 4);
  float* joints  = (float*)alloc((size_t)B * 42 * 4);
  float* grouped = (float*)alloc((size_t)B * 896 * 6 * 4);
  int*   fidx3   = (int*)alloc((size_t)B * 512 * 4);
  float* newxyz3 = (float*)alloc((size_t)B * 512 * 3 * 4);
  int*   gidx3   = (int*)alloc((size_t)B * 512 * 64 * 4);
  float* l3_pts  = (float*)alloc((size_t)B * 128 * 512 * 4);
  float* h3      = (float*)alloc((size_t)B * 256 * 4);
  // padded bf16 SA weights [Nout][Kpad]
  bf16_t* s1w0p = (bf16_t*)alloc((size_t)64 * 32 * 2);
  bf16_t* s1w1p = (bf16_t*)alloc((size_t)64 * 64 * 2);
  bf16_t* s1w2p = (bf16_t*)alloc((size_t)128 * 64 * 2);
  bf16_t* s2w0p = (bf16_t*)alloc((size_t)128 * 160 * 2);
  bf16_t* s2w1p = (bf16_t*)alloc((size_t)128 * 128 * 2);
  bf16_t* s2w2p = (bf16_t*)alloc((size_t)256 * 128 * 2);
  bf16_t* s3w0p = (bf16_t*)alloc((size_t)64 * 32 * 2);
  bf16_t* s3w1p = (bf16_t*)alloc((size_t)64 * 64 * 2);
  bf16_t* s3w2p = (bf16_t*)alloc((size_t)128 * 64 * 2);

  // ---- pre-pad all SA weights to bf16 (one-shot, tiny) ----
  auto pad = [&](const float* w, int Kc, int Kpad, int Nout, bf16_t* dst) {
    const int tot = Nout * Kpad;
    pad_weights_kernel<<<(tot + 255) / 256, 256, 0, stream>>>(w, Kc, Kpad, tot,
                                                              dst);
  };
  pad(s1w0, 6, 32, 64, s1w0p);   pad(s1w1, 64, 64, 64, s1w1p);
  pad(s1w2, 64, 64, 128, s1w2p);
  pad(s2w0, 131, 160, 128, s2w0p); pad(s2w1, 128, 128, 128, s2w1p);
  pad(s2w2, 128, 128, 256, s2w2p);
  pad(s3w0, 6, 32, 64, s3w0p);   pad(s3w1, 64, 64, 64, s3w1p);
  pad(s3w2, 64, 64, 128, s3w2p);

  // ======================= SA1 (N=1024 -> S=512) ===========================
  fps_kernel<<<B, 1024, 0, stream>>>(xyz6, 6, 1024, 512, fidx1);
  {
    const int tot = B * 512;
    gather_newxyz_kernel<<<(tot + 255) / 256, 256, 0, stream>>>(
        xyz6, 6, 1024, fidx1, tot, newxyz1, 512);
    ball_query_kernel<<<(tot + 255) / 256, 256, 0, stream>>>(
        xyz6, 6, 1024, newxyz1, tot, 512, 0.2f * 0.2f, gidx1);
  }
  {
    const int Cpad0 = 64, Cpad1 = 128;
    const size_t shmem = (size_t)64 * (Cpad0 + Cpad1) * sizeof(bf16_t);
    sa_mlp_kernel<<<B * 512, 128, shmem, stream>>>(
        xyz6, 6, 1024, xyz6 + 3, 0, 6, 3, newxyz1, gidx1, 512, 6, Cpad0, Cpad1,
        s1w0p, 32, s1b0, s1g0, s1be0, s1m0, s1v0, 64,
        s1w1p, s1b1, s1g1, s1be1, s1m1, s1v1, 64,
        s1w2p, s1b2, s1g2, s1be2, s1m2, s1v2, 128, l1_pts);
  }
  stage_gemm_kernel<<<16, 256, 0, stream>>>(l1_pts, 128 * 512, st1l1w, 256,
                                            st1l1b, st1g, st1be, st1m, st1v,
                                            0, h1);
  stage_gemm_kernel<<<3, 256, 0, stream>>>(h1, 256, st1l2w, 42, st1l2b,
                                           st1g, st1be, st1m, st1v, 1, e1);

  // ======================= SA2 (N=512 -> S=128) ============================
  fps_kernel<<<B, 1024, 0, stream>>>(newxyz1, 3, 512, 128, fidx2);
  {
    const int tot = B * 128;
    gather_newxyz_kernel<<<(tot + 255) / 256, 256, 0, stream>>>(
        newxyz1, 3, 512, fidx2, tot, newxyz2, 128);
    ball_query_kernel<<<(tot + 255) / 256, 256, 0, stream>>>(
        newxyz1, 3, 512, newxyz2, tot, 128, 0.5f * 0.5f, gidx2);
  }
  {
    const int Cpad0 = 160, Cpad1 = 256;
    const size_t shmem = (size_t)64 * (Cpad0 + Cpad1) * sizeof(bf16_t);
    sa_mlp_kernel<<<B * 128, 128, shmem, stream>>>(
        newxyz1, 3, 512, l1_pts, 1, 0, 128, newxyz2, gidx2, 128, 131,
        Cpad0, Cpad1,
        s2w0p, 160, s2b0, s2g0, s2be0, s2m0, s2v0, 128,
        s2w1p, s2b1, s2g1, s2be1, s2m1, s2v1, 128,
        s2w2p, s2b2, s2g2, s2be2, s2m2, s2v2, 256, l2_pts);
  }
  stage_gemm_kernel<<<8, 256, 0, stream>>>(l2_pts, 256 * 128, st2l1w, 128,
                                           st2l1b, st2g, st2be, st2m, st2v,
                                           0, h2);
  stage_gemm_kernel<<<3, 256, 0, stream>>>(h2, 128, st2l2w, 42, st2l2b,
                                           st2g, st2be, st2m, st2v, 1, e2);

  // ============== joints + kNN regrouping around joints ====================
  joints_kernel<<<3, 256, 0, stream>>>(e2, pca_mean, pca_coeff, joints);
  topk_group_kernel<<<B * 14, 256, 0, stream>>>(xyz6, joints, grouped);

  // ======================= SA3 (N=896 -> S=512) ============================
  fps_kernel<<<B, 1024, 0, stream>>>(grouped, 6, 896, 512, fidx3);
  {
    const int tot = B * 512;
    gather_newxyz_kernel<<<(tot + 255) / 256, 256, 0, stream>>>(
        grouped, 6, 896, fidx3, tot, newxyz3, 512);
    ball_query_kernel<<<(tot + 255) / 256, 256, 0, stream>>>(
        grouped, 6, 896, newxyz3, tot, 512, 1.0f * 1.0f, gidx3);
  }
  {
    const int Cpad0 = 64, Cpad1 = 128;
    const size_t shmem = (size_t)64 * (Cpad0 + Cpad1) * sizeof(bf16_t);
    sa_mlp_kernel<<<B * 512, 128, shmem, stream>>>(
        grouped, 6, 896, grouped + 3, 0, 6, 3, newxyz3, gidx3, 512, 6,
        Cpad0, Cpad1,
        s3w0p, 32, s3b0, s3g0, s3be0, s3m0, s3v0, 64,
        s3w1p, s3b1, s3g1, s3be1, s3m1, s3v1, 64,
        s3w2p, s3b2, s3g2, s3be2, s3m2, s3v2, 128, l3_pts);
  }
  stage_gemm_kernel<<<16, 256, 0, stream>>>(l3_pts, 128 * 512, st1l1w, 256,
                                            st1l1b, st1g, st1be, st1m, st1v,
                                            0, h3);
  stage_gemm_kernel<<<3, 256, 0, stream>>>(h3, 256, st1l2w, 42, st1l2b,
                                           st1g, st1be, st1m, st1v, 1, e3);
}
